// ShapeRetrieval_68118181314987
// MI455X (gfx1250) — matrix operand order repaired and checked by reading, not compiled, not run on hardware
//
#include <hip/hip_runtime.h>

typedef __attribute__((ext_vector_type(16))) __bf16 v16bf;
typedef __attribute__((ext_vector_type(8)))  float  v8f;

#define DIM 256
#define TN  64
#define TM  128
#define KB  32
#define LDA 34
#define LDB 34

__device__ __forceinline__ unsigned short bf16_rn(float x) {
    unsigned int u = __float_as_uint(x);
    unsigned int r = u + 0x7FFFu + ((u >> 16) & 1u);   // RNE to bf16
    return (unsigned short)(r >> 16);
}
__device__ __forceinline__ float bf16_to_f(unsigned short h) {
    return __uint_as_float(((unsigned int)h) << 16);
}
__device__ __forceinline__ unsigned long long u64max(unsigned long long a, unsigned long long b) {
    return a > b ? a : b;
}

union FragU {
    uint4 q[2];
    unsigned int u[8];
    v16bf v;
};

// ---------------------------------------------------------------- init
__global__ void init_kernel(unsigned long long* best, int N) {
    int n = blockIdx.x * blockDim.x + threadIdx.x;
    if (n < N) best[n] = 0ull;
}

// ---------------- gallery: fused inv-norm + bf16 hi/lo split (one wave/row)
__global__ __launch_bounds__(256) void prep_g_kernel(const float* __restrict__ G,
                                                     unsigned short* __restrict__ Ghi,
                                                     unsigned short* __restrict__ Glo,
                                                     float* __restrict__ invn, int M) {
    const int lane = threadIdx.x & 31;
    const int row  = (int)((blockIdx.x * blockDim.x + threadIdx.x) >> 5);
    if (row >= M) return;
    const float* gp = G + (size_t)row * DIM + lane * 8;
    unsigned short* hp = Ghi + (size_t)row * DIM + lane * 8;
    unsigned short* lp = Glo + (size_t)row * DIM + lane * 8;
    float s = 0.0f;
#pragma unroll
    for (int e = 0; e < 8; e++) {
        float x = gp[e];
        s += x * x;
        unsigned short h = bf16_rn(x);
        hp[e] = h;
        lp[e] = bf16_rn(x - bf16_to_f(h));
    }
#pragma unroll
    for (int m = 1; m < 32; m <<= 1) s += __shfl_xor(s, m, 32);
    if (lane == 0) invn[row] = 1.0f / fmaxf(sqrtf(s), 1e-12f);
}

// ---------------- queries: bf16 hi/lo split
__global__ __launch_bounds__(256) void prep_q_kernel(const float* __restrict__ Q,
                                                     unsigned short* __restrict__ Qhi,
                                                     unsigned short* __restrict__ Qlo, int total) {
    int i = blockIdx.x * blockDim.x + threadIdx.x;
    if (i < total) {
        float x = Q[i];
        unsigned short h = bf16_rn(x);
        Qhi[i] = h;
        Qlo[i] = bf16_rn(x - bf16_to_f(h));
    }
}

// ------- gallery inverse norms only (fallback path)
__global__ __launch_bounds__(256) void norm_kernel(const float* __restrict__ G,
                                                   float* __restrict__ invn, int M) {
    const int lane = threadIdx.x & 31;
    const int row  = (int)((blockIdx.x * blockDim.x + threadIdx.x) >> 5);
    if (row >= M) return;
    const float* gp = G + (size_t)row * DIM + lane * 8;
    float s = 0.0f;
#pragma unroll
    for (int e = 0; e < 8; e++) { float x = gp[e]; s += x * x; }
#pragma unroll
    for (int m = 1; m < 32; m <<= 1) s += __shfl_xor(s, m, 32);
    if (lane == 0) invn[row] = 1.0f / fmaxf(sqrtf(s), 1e-12f);
}

// ================= FAST PATH: direct-global-fragment split-bf16 GEMM + argmax
// A fragment (16-bit A 16x32 ISA layout): lane holds row, VGPR0-3 = K 8g..8g+7,
// VGPR4-7 = K 16+8g..23+8g  -> two contiguous 16B chunks of a row-major row.
// B fragment (32x16): lane holds col n=lane%16, VGPR0-3 = K 8g..8g+7,
// VGPR4-7 = K 16+8g..23+8g  -> two 16B chunks of a row-major gallery row.
__global__ __launch_bounds__(256) void gemm_argmax_fast(
    const unsigned short* __restrict__ Qhi, const unsigned short* __restrict__ Qlo,
    const unsigned short* __restrict__ Ghi, const unsigned short* __restrict__ Glo,
    const float* __restrict__ invn, unsigned long long* __restrict__ best, int M)
{
    __shared__ unsigned long long sBest[TN];

    const int tid   = threadIdx.x;
    const int lane  = tid & 31;
    const int wid   = tid >> 5;
    const int wr    = wid >> 2;          // 0..1
    const int wc    = wid & 3;           // 0..3
    const int half_ = lane >> 4;         // 0/1
    const int l16   = lane & 15;
    const int nBase = blockIdx.y * TN;
    const int mBase = blockIdx.x * TM;

    if (tid < TN) sBest[tid] = 0ull;
    __syncthreads();

    const int aRow0 = nBase + wr * 32 + l16;
    const int bRow0 = mBase + wc * 32 + l16;
    const int bRowC[2] = { min(bRow0, M - 1), min(bRow0 + 16, M - 1) };

    v8f acc[2][2] = {};

    for (int k0 = 0; k0 < DIM; k0 += KB) {
        FragU aHi[2], aLo[2], bHi[2], bLo[2];
#pragma unroll
        for (int i = 0; i < 2; i++) {
            const size_t base = (size_t)(aRow0 + 16 * i) * DIM + k0 + 8 * half_;
            aHi[i].q[0] = *(const uint4*)(Qhi + base);
            aHi[i].q[1] = *(const uint4*)(Qhi + base + 16);
            aLo[i].q[0] = *(const uint4*)(Qlo + base);
            aLo[i].q[1] = *(const uint4*)(Qlo + base + 16);
        }
#pragma unroll
        for (int j = 0; j < 2; j++) {
            const size_t base = (size_t)bRowC[j] * DIM + k0 + 8 * half_;
            bHi[j].q[0] = *(const uint4*)(Ghi + base);
            bHi[j].q[1] = *(const uint4*)(Ghi + base + 16);
            bLo[j].q[0] = *(const uint4*)(Glo + base);
            bLo[j].q[1] = *(const uint4*)(Glo + base + 16);
        }
#pragma unroll
        for (int i = 0; i < 2; i++)
#pragma unroll
            for (int j = 0; j < 2; j++) {
                acc[i][j] = __builtin_amdgcn_wmma_f32_16x16x32_bf16(
                    false, aHi[i].v, false, bHi[j].v, (short)0, acc[i][j], false, false);
                acc[i][j] = __builtin_amdgcn_wmma_f32_16x16x32_bf16(
                    false, aHi[i].v, false, bLo[j].v, (short)0, acc[i][j], false, false);
                acc[i][j] = __builtin_amdgcn_wmma_f32_16x16x32_bf16(
                    false, aLo[i].v, false, bHi[j].v, (short)0, acc[i][j], false, false);
            }
    }

#pragma unroll
    for (int i = 0; i < 2; i++) {
#pragma unroll
        for (int r = 0; r < 8; r++) {
            const int qLocal = wr * 32 + 16 * i + r + (half_ << 3);
            unsigned long long pack = 0ull;
#pragma unroll
            for (int j = 0; j < 2; j++) {
                const int mIdx = mBase + wc * 32 + 16 * j + l16;
                if (mIdx < M) {
                    float s = acc[i][j][r] * invn[mIdx];
                    unsigned int b = __float_as_uint(s);
                    unsigned int key = (b & 0x80000000u) ? ~b : (b | 0x80000000u);
                    unsigned long long p = ((unsigned long long)key << 32)
                                         | (unsigned long long)(0xFFFFFFFFu - (unsigned int)mIdx);
                    pack = u64max(pack, p);
                }
            }
#pragma unroll
            for (int m = 1; m <= 8; m <<= 1)
                pack = u64max(pack, (unsigned long long)__shfl_xor(pack, m, 32));
            if (l16 == 0) atomicMax(&sBest[qLocal], pack);
        }
    }
    __syncthreads();
    if (tid < TN) {
        unsigned long long p = sBest[tid];
        if (p) atomicMax(&best[nBase + tid], p);
    }
}

// ================= FALLBACK: LDS-staged split-bf16 GEMM + argmax (small ws)
__global__ __launch_bounds__(256) void gemm_argmax_kernel(
    const float* __restrict__ Q, const float* __restrict__ G,
    const float* __restrict__ invn, unsigned long long* __restrict__ best, int M)
{
    __shared__ unsigned short sAhi[TN * LDA];
    __shared__ unsigned short sAlo[TN * LDA];
    __shared__ unsigned short sBhi[TM * LDB];
    __shared__ unsigned short sBlo[TM * LDB];
    __shared__ unsigned long long sBest[TN];

    const int tid   = threadIdx.x;
    const int lane  = tid & 31;
    const int wid   = tid >> 5;
    const int wr    = wid >> 2;
    const int wc    = wid & 3;
    const int half_ = lane >> 4;
    const int nBase = blockIdx.y * TN;
    const int mBase = blockIdx.x * TM;

    if (tid < TN) sBest[tid] = 0ull;

    v8f acc[2][2] = {};

    const int aRow = tid >> 2;
    const int aCol = (tid & 3) << 3;
    const int bRow = tid >> 1;
    const int bCol = (tid & 1) << 4;
    const bool bValid = (mBase + bRow) < M;
    const float* qRow = Q + (size_t)(nBase + aRow) * DIM + aCol;
    const float* gRow = G + (size_t)(mBase + bRow) * DIM + bCol;

    for (int k0 = 0; k0 < DIM; k0 += KB) {
        __syncthreads();
#pragma unroll
        for (int e = 0; e < 8; e++) {
            float x = qRow[k0 + e];
            unsigned short hi = bf16_rn(x);
            sAhi[aRow * LDA + aCol + e] = hi;
            sAlo[aRow * LDA + aCol + e] = bf16_rn(x - bf16_to_f(hi));
        }
#pragma unroll
        for (int e = 0; e < 16; e++) {
            float x = bValid ? gRow[k0 + e] : 0.0f;
            unsigned short hi = bf16_rn(x);
            sBhi[bRow * LDB + bCol + e] = hi;
            sBlo[bRow * LDB + bCol + e] = bf16_rn(x - bf16_to_f(hi));
        }
        __syncthreads();

        FragU aHi[2], aLo[2], bHi[2], bLo[2];
        const int rA0 = wr * 32 + (lane & 15);
#pragma unroll
        for (int i = 0; i < 2; i++) {
            const int row = rA0 + 16 * i;
#pragma unroll
            for (int v = 0; v < 8; v++) {
                const int kb = ((v & 4) << 2) + (half_ << 3) + ((v & 3) << 1);
                aHi[i].u[v] = *(const unsigned int*)&sAhi[row * LDA + kb];
                aLo[i].u[v] = *(const unsigned int*)&sAlo[row * LDA + kb];
            }
        }
#pragma unroll
        for (int j = 0; j < 2; j++) {
            const int n = wc * 32 + 16 * j + (lane & 15);
#pragma unroll
            for (int v = 0; v < 8; v++) {
                const int kb = ((v & 4) << 2) + (half_ << 3) + ((v & 3) << 1);
                bHi[j].u[v] = *(const unsigned int*)&sBhi[n * LDB + kb];
                bLo[j].u[v] = *(const unsigned int*)&sBlo[n * LDB + kb];
            }
        }
#pragma unroll
        for (int i = 0; i < 2; i++)
#pragma unroll
            for (int j = 0; j < 2; j++) {
                acc[i][j] = __builtin_amdgcn_wmma_f32_16x16x32_bf16(
                    false, aHi[i].v, false, bHi[j].v, (short)0, acc[i][j], false, false);
                acc[i][j] = __builtin_amdgcn_wmma_f32_16x16x32_bf16(
                    false, aHi[i].v, false, bLo[j].v, (short)0, acc[i][j], false, false);
                acc[i][j] = __builtin_amdgcn_wmma_f32_16x16x32_bf16(
                    false, aLo[i].v, false, bHi[j].v, (short)0, acc[i][j], false, false);
            }
    }

#pragma unroll
    for (int i = 0; i < 2; i++) {
#pragma unroll
        for (int r = 0; r < 8; r++) {
            const int qLocal = wr * 32 + 16 * i + r + (half_ << 3);
            unsigned long long pack = 0ull;
#pragma unroll
            for (int j = 0; j < 2; j++) {
                const int mIdx = mBase + wc * 32 + 16 * j + (lane & 15);
                if (mIdx < M) {
                    float s = acc[i][j][r] * invn[mIdx];
                    unsigned int b = __float_as_uint(s);
                    unsigned int key = (b & 0x80000000u) ? ~b : (b | 0x80000000u);
                    unsigned long long p = ((unsigned long long)key << 32)
                                         | (unsigned long long)(0xFFFFFFFFu - (unsigned int)mIdx);
                    pack = u64max(pack, p);
                }
            }
#pragma unroll
            for (int m = 1; m <= 8; m <<= 1)
                pack = u64max(pack, (unsigned long long)__shfl_xor(pack, m, 32));
            if ((lane & 15) == 0) atomicMax(&sBest[qLocal], pack);
        }
    }
    __syncthreads();
    if (tid < TN) {
        unsigned long long p = sBest[tid];
        if (p) atomicMax(&best[nBase + tid], p);
    }
}

// ---------------------------------------------------------------- gather
__global__ void finalize_kernel(const unsigned long long* __restrict__ best,
                                const int* __restrict__ cat, const int* __restrict__ shp,
                                float* __restrict__ out, int N) {
    int n = blockIdx.x * blockDim.x + threadIdx.x;
    if (n < N) {
        unsigned long long p = best[n];
        unsigned int m = 0xFFFFFFFFu - (unsigned int)(p & 0xFFFFFFFFull);
        out[n]     = (float)cat[m];
        out[N + n] = (float)shp[m];
    }
}

static inline size_t align256(size_t x) { return (x + 255) & ~(size_t)255; }

extern "C" void kernel_launch(void* const* d_in, const int* in_sizes, int n_in,
                              void* d_out, int out_size, void* d_ws, size_t ws_size,
                              hipStream_t stream) {
    const float* Q  = (const float*)d_in[0];   // [N, 256]
    const float* G  = (const float*)d_in[1];   // [M, 256]
    const int* cat  = (const int*)d_in[2];     // [M]
    const int* shp  = (const int*)d_in[3];     // [M]
    const int  N    = in_sizes[0] / DIM;       // 2048
    const int  M    = in_sizes[2];             // 100000
    float* out      = (float*)d_out;

    char* ws = (char*)d_ws;
    size_t off = 0;
    const size_t bestOff = off; off = align256(off + (size_t)N * 8);
    const size_t invnOff = off; off = align256(off + (size_t)M * 4);
    const size_t smallNeed = off;
    const size_t qhiOff = off;  off = align256(off + (size_t)N * DIM * 2);
    const size_t qloOff = off;  off = align256(off + (size_t)N * DIM * 2);
    const size_t ghiOff = off;  off = align256(off + (size_t)M * DIM * 2);
    const size_t gloOff = off;  off = align256(off + (size_t)M * DIM * 2);
    const size_t bigNeed = off;

    unsigned long long* best = (unsigned long long*)(ws + bestOff);
    float* invn = (float*)(ws + invnOff);

    init_kernel<<<(N + 255) / 256, 256, 0, stream>>>(best, N);

    dim3 grid((M + TM - 1) / TM, (N + TN - 1) / TN);

    if (ws_size >= bigNeed) {
        unsigned short* Qhi = (unsigned short*)(ws + qhiOff);
        unsigned short* Qlo = (unsigned short*)(ws + qloOff);
        unsigned short* Ghi = (unsigned short*)(ws + ghiOff);
        unsigned short* Glo = (unsigned short*)(ws + gloOff);

        prep_g_kernel<<<(M + 7) / 8, 256, 0, stream>>>(G, Ghi, Glo, invn, M);
        prep_q_kernel<<<(N * DIM + 255) / 256, 256, 0, stream>>>(Q, Qhi, Qlo, N * DIM);
        gemm_argmax_fast<<<grid, 256, 0, stream>>>(Qhi, Qlo, Ghi, Glo, invn, best, M);
    } else if (ws_size >= smallNeed) {
        norm_kernel<<<(M + 7) / 8, 256, 0, stream>>>(G, invn, M);
        gemm_argmax_kernel<<<grid, 256, 0, stream>>>(Q, G, invn, best, M);
    }

    finalize_kernel<<<(N + 255) / 256, 256, 0, stream>>>(best, cat, shp, out, N);
}
